// Encoder_Decoder_61667140436424
// MI455X (gfx1250) — matrix operand order repaired and checked by reading, not compiled, not run on hardware
//
#include <hip/hip_runtime.h>
#include <math.h>

#define SRC_LEN   48
#define TGT_LEN   48
#define BATCH     64
#define HID       512
#define VOCAB     32000
#define GATES     (4 * HID)        // 2048
#define DEC_STEPS (TGT_LEN - 1)    // 47

typedef __attribute__((ext_vector_type(2)))  float   v2f;
typedef __attribute__((ext_vector_type(8)))  float   v8f;
typedef __attribute__((ext_vector_type(8)))  __bf16  v8bf;
typedef __attribute__((ext_vector_type(16))) __bf16  v16bf;

// ---------------------------------------------------------------------------
// Convert W_out (f32 [VOCAB][HID]) to bf16 once per call (stays in L2).
// ---------------------------------------------------------------------------
__global__ void w_to_bf16_kernel(const float* __restrict__ in,
                                 __bf16* __restrict__ out, int n4) {
  int i = blockIdx.x * blockDim.x + threadIdx.x;
  if (i < n4) {
    const float4 v = reinterpret_cast<const float4*>(in)[i];
    __bf16 o0 = (__bf16)v.x, o1 = (__bf16)v.y, o2 = (__bf16)v.z, o3 = (__bf16)v.w;
    out[4 * i + 0] = o0; out[4 * i + 1] = o1;
    out[4 * i + 2] = o2; out[4 * i + 3] = o3;
  }
}

// ---------------------------------------------------------------------------
// Embedding gather with row-0 zeroing (emb.at[0].set(0)).
// X[t][b][h] = (tok==0) ? 0 : emb[tok][h]
// ---------------------------------------------------------------------------
__global__ void embed_kernel(const int* __restrict__ toks,
                             const float* __restrict__ emb,
                             float* __restrict__ X, int steps) {
  int idx = blockIdx.x * blockDim.x + threadIdx.x;
  int total = steps * BATCH * HID;
  if (idx >= total) return;
  int t = idx / (BATCH * HID);
  int r = idx - t * (BATCH * HID);
  int b = r >> 9;           // / HID
  int h = r & (HID - 1);
  int tok = toks[t * BATCH + b];
  X[idx] = (tok == 0) ? 0.0f : emb[(size_t)tok * HID + h];
}

__global__ void zero_kernel(float* __restrict__ p, int n) {
  int i = blockIdx.x * blockDim.x + threadIdx.x;
  if (i < n) p[i] = 0.0f;
}

// ---------------------------------------------------------------------------
// Gates GEMM (fp32 WMMA 16x16x4):  G[64][2048] = X@W_ih^T + H@W_hh^T
// All matrices row-major with K=512 contiguous. 8 waves/block; each wave
// owns one 16-wide N tile and all 4 M tiles (B fragment reused 4x).
// ---------------------------------------------------------------------------
__global__ __launch_bounds__(256)
void gates_wmma_kernel(const float* __restrict__ X, const float* __restrict__ H,
                       const float* __restrict__ Wih, const float* __restrict__ Whh,
                       float* __restrict__ G) {
  const int wave = threadIdx.x >> 5;
  const int lane = threadIdx.x & 31;
  const int n0   = (blockIdx.x * 8 + wave) * 16;   // N tile base (0..2032)
  const int lrow = lane & 15;
  const int kh   = (lane < 16) ? 0 : 2;            // f32 frag K half select

  v8f acc[4] = {v8f{}, v8f{}, v8f{}, v8f{}};

  // X @ W_ih^T
  {
    const float* wb = Wih + (size_t)(n0 + lrow) * HID + kh;
    const float* xb = X   + (size_t)lrow * HID + kh;
    for (int k = 0; k < HID; k += 4) {
      v2f b = *(const v2f*)(wb + k);
#pragma unroll
      for (int mt = 0; mt < 4; ++mt) {
        v2f a = *(const v2f*)(xb + (size_t)mt * 16 * HID + k);
        acc[mt] = __builtin_amdgcn_wmma_f32_16x16x4_f32(
            false, a, false, b, (short)0, acc[mt], false, false);
      }
    }
  }
  // H @ W_hh^T (accumulate)
  {
    const float* wb = Whh + (size_t)(n0 + lrow) * HID + kh;
    const float* hb = H   + (size_t)lrow * HID + kh;
    for (int k = 0; k < HID; k += 4) {
      v2f b = *(const v2f*)(wb + k);
#pragma unroll
      for (int mt = 0; mt < 4; ++mt) {
        v2f a = *(const v2f*)(hb + (size_t)mt * 16 * HID + k);
        acc[mt] = __builtin_amdgcn_wmma_f32_16x16x4_f32(
            false, a, false, b, (short)0, acc[mt], false, false);
      }
    }
  }
  // C/D layout: VGPR v -> M = mt*16 + (lane<16 ? v : v+8), N = n0 + lane%16
#pragma unroll
  for (int mt = 0; mt < 4; ++mt) {
#pragma unroll
    for (int v = 0; v < 8; ++v) {
      int M = mt * 16 + ((lane < 16) ? v : v + 8);
      G[(size_t)M * GATES + n0 + lrow] = acc[mt][v];
    }
  }
}

// ---------------------------------------------------------------------------
// Fused LSTM cell update (biases + activations + optional pad mask).
// Encoder passes toks (pad where tok==0); decoder passes h_bf to export
// bf16 hidden state for the logits WMMA.
// ---------------------------------------------------------------------------
__device__ __forceinline__ float sigm(float x) { return 1.0f / (1.0f + expf(-x)); }

__global__ __launch_bounds__(256)
void lstm_cell_kernel(const float* __restrict__ G,
                      float* __restrict__ h, float* __restrict__ c,
                      const float* __restrict__ b_ih, const float* __restrict__ b_hh,
                      const int* __restrict__ toks,   // nullable (encoder pad mask)
                      __bf16* __restrict__ h_bf) {    // nullable (decoder export)
  int idx = blockIdx.x * blockDim.x + threadIdx.x;    // < BATCH*HID
  int m = idx >> 9;
  int n = idx & (HID - 1);
  const float* gr = G + (size_t)m * GATES;
  float gi = gr[n]            + b_ih[n]            + b_hh[n];
  float gf = gr[HID + n]      + b_ih[HID + n]      + b_hh[HID + n];
  float gg = gr[2 * HID + n]  + b_ih[2 * HID + n]  + b_hh[2 * HID + n];
  float go = gr[3 * HID + n]  + b_ih[3 * HID + n]  + b_hh[3 * HID + n];

  float cPrev = c[idx];
  float c2 = sigm(gf) * cPrev + sigm(gi) * tanhf(gg);
  float h2 = sigm(go) * tanhf(c2);
  if (toks != nullptr && toks[m] == 0) { c2 = cPrev; h2 = h[idx]; }
  h[idx] = h2;
  c[idx] = c2;
  if (h_bf != nullptr) h_bf[idx] = (__bf16)h2;
}

// ---------------------------------------------------------------------------
// Logits GEMM (bf16 WMMA 16x16x32, f32 accumulate):
//   logits[64][32000] = h_bf[64][512] @ Wout_bf[32000][512]^T
// 8 waves/block -> 128-wide N slab; wave = one N tile x 4 M tiles.
// ---------------------------------------------------------------------------
__global__ __launch_bounds__(256)
void logits_wmma_kernel(const __bf16* __restrict__ A,   // [64][512]
                        const __bf16* __restrict__ W,   // [32000][512]
                        float* __restrict__ C) {        // [64][32000]
  const int wave = threadIdx.x >> 5;
  const int lane = threadIdx.x & 31;
  const int n0   = (blockIdx.x * 8 + wave) * 16;
  const int lrow = lane & 15;
  const int kkB  = (lane < 16) ? 0 : 16;   // bf16 B frag: contiguous K group
  const int kkA  = (lane < 16) ? 0 : 16;   // bf16 A frag: contiguous K group

  v8f acc[4] = {v8f{}, v8f{}, v8f{}, v8f{}};
  const __bf16* wb = W + (size_t)(n0 + lrow) * HID + kkB;

  for (int k = 0; k < HID; k += 32) {
    v16bf b = *(const v16bf*)(wb + k);     // W[n][k+kkB .. k+kkB+15], 32B aligned
#pragma unroll
    for (int mt = 0; mt < 4; ++mt) {
      const __bf16* ap = A + (size_t)(mt * 16 + lrow) * HID + k + kkA;
      v8bf lo = *(const v8bf*)(ap);        // K = k+kkA   .. +7
      v8bf hi = *(const v8bf*)(ap + 8);    // K = k+8+kkA .. +7
      v16bf a = __builtin_shufflevector(lo, hi, 0, 1, 2, 3, 4, 5, 6, 7,
                                        8, 9, 10, 11, 12, 13, 14, 15);
      acc[mt] = __builtin_amdgcn_wmma_f32_16x16x32_bf16(
          false, a, false, b, (short)0, acc[mt], false, false);
    }
  }
#pragma unroll
  for (int mt = 0; mt < 4; ++mt) {
#pragma unroll
    for (int v = 0; v < 8; ++v) {
      int M = mt * 16 + ((lane < 16) ? v : v + 8);
      C[(size_t)M * VOCAB + n0 + lrow] = acc[mt][v];
    }
  }
}

// ---------------------------------------------------------------------------
// Per-row online logsumexp + NLL contribution. One block per batch row.
// Fixed-order tree reduction => deterministic across graph replays.
// ---------------------------------------------------------------------------
__global__ __launch_bounds__(256)
void row_loss_kernel(const float* __restrict__ logits,
                     const float* __restrict__ b_out,
                     const int* __restrict__ tok_next,   // target_lines+(t+1)*BATCH
                     float* __restrict__ loss_part) {    // + t*BATCH
  __shared__ float smax[256];
  __shared__ float ssum[256];
  const int m   = blockIdx.x;
  const int tid = threadIdx.x;
  const float* row = logits + (size_t)m * VOCAB;

  float lmax = -INFINITY, lsum = 0.0f;
  for (int n = tid; n < VOCAB; n += 256) {
    float v = row[n] + b_out[n];
    if (v > lmax) { lsum = lsum * expf(lmax - v) + 1.0f; lmax = v; }
    else          { lsum += expf(v - lmax); }
  }
  smax[tid] = lmax; ssum[tid] = lsum;
  __syncthreads();
  for (int s = 128; s > 0; s >>= 1) {
    if (tid < s) {
      float m1 = smax[tid], s1 = ssum[tid];
      float m2 = smax[tid + s], s2 = ssum[tid + s];
      float M = fmaxf(m1, m2);
      ssum[tid] = s1 * expf(m1 - M) + s2 * expf(m2 - M);
      smax[tid] = M;
    }
    __syncthreads();
  }
  if (tid == 0) {
    int tn = tok_next[m];
    float tl = row[tn] + b_out[tn];
    loss_part[m] = (smax[0] + logf(ssum[0])) - tl;
  }
}

__global__ __launch_bounds__(256)
void final_loss_kernel(const float* __restrict__ loss_part, float* __restrict__ out) {
  __shared__ float s[256];
  float acc = 0.0f;
  for (int i = threadIdx.x; i < DEC_STEPS * BATCH; i += 256) acc += loss_part[i];
  s[threadIdx.x] = acc;
  __syncthreads();
  for (int st = 128; st > 0; st >>= 1) {
    if (threadIdx.x < st) s[threadIdx.x] += s[threadIdx.x + st];
    __syncthreads();
  }
  if (threadIdx.x == 0) out[0] = s[0] * (1.0f / BATCH);
}

// ---------------------------------------------------------------------------
extern "C" void kernel_launch(void* const* d_in, const int* in_sizes, int n_in,
                              void* d_out, int out_size, void* d_ws, size_t ws_size,
                              hipStream_t stream) {
  (void)in_sizes; (void)n_in; (void)out_size; (void)ws_size;
  const int*   input_lines  = (const int*)d_in[0];
  const int*   target_lines = (const int*)d_in[1];
  const float* emb_in  = (const float*)d_in[2];
  const float* emb_tgt = (const float*)d_in[3];
  const float* W_ih_e  = (const float*)d_in[4];
  const float* W_hh_e  = (const float*)d_in[5];
  const float* b_ih_e  = (const float*)d_in[6];
  const float* b_hh_e  = (const float*)d_in[7];
  const float* W_ih_d  = (const float*)d_in[8];
  const float* W_hh_d  = (const float*)d_in[9];
  const float* b_ih_d  = (const float*)d_in[10];
  const float* b_hh_d  = (const float*)d_in[11];
  const float* W_out   = (const float*)d_in[12];
  const float* b_out   = (const float*)d_in[13];

  // Workspace carve-up (256B aligned slices).
  char*  ws  = (char*)d_ws;
  size_t off = 0;
  auto carve = [&](size_t bytes) -> void* {
    void* p = ws + off;
    off += (bytes + 255) & ~(size_t)255;
    return p;
  };
  __bf16* Wout_bf   = (__bf16*)carve((size_t)VOCAB * HID * sizeof(__bf16));
  float*  Xe        = (float*) carve((size_t)SRC_LEN * BATCH * HID * sizeof(float));
  float*  Xd        = (float*) carve((size_t)DEC_STEPS * BATCH * HID * sizeof(float));
  float*  G         = (float*) carve((size_t)BATCH * GATES * sizeof(float));
  float*  hc        = (float*) carve((size_t)2 * BATCH * HID * sizeof(float));
  float*  h         = hc;
  float*  c         = hc + BATCH * HID;
  __bf16* h_bf      = (__bf16*)carve((size_t)BATCH * HID * sizeof(__bf16));
  float*  logits    = (float*) carve((size_t)BATCH * VOCAB * sizeof(float));
  float*  loss_part = (float*) carve((size_t)DEC_STEPS * BATCH * sizeof(float));

  // Setup: bf16 weights, embeddings, zero state.
  {
    int n4 = VOCAB * HID / 4;
    w_to_bf16_kernel<<<(n4 + 255) / 256, 256, 0, stream>>>(W_out, Wout_bf, n4);
    int ne = SRC_LEN * BATCH * HID;
    embed_kernel<<<(ne + 255) / 256, 256, 0, stream>>>(input_lines, emb_in, Xe, SRC_LEN);
    int nd = DEC_STEPS * BATCH * HID;
    embed_kernel<<<(nd + 255) / 256, 256, 0, stream>>>(target_lines, emb_tgt, Xd, DEC_STEPS);
    int nz = 2 * BATCH * HID;
    zero_kernel<<<(nz + 255) / 256, 256, 0, stream>>>(hc, nz);
  }

  const int gatesGrid = GATES / 128;          // 16 blocks (8 N-tiles each)
  const int cellGrid  = (BATCH * HID) / 256;  // 128 blocks
  const int logitGrid = VOCAB / 128;          // 250 blocks

  // Encoder recurrence.
  for (int t = 0; t < SRC_LEN; ++t) {
    gates_wmma_kernel<<<gatesGrid, 256, 0, stream>>>(
        Xe + (size_t)t * BATCH * HID, h, W_ih_e, W_hh_e, G);
    lstm_cell_kernel<<<cellGrid, 256, 0, stream>>>(
        G, h, c, b_ih_e, b_hh_e, input_lines + t * BATCH, nullptr);
  }

  // Decoder recurrence + projection + loss.
  for (int t = 0; t < DEC_STEPS; ++t) {
    gates_wmma_kernel<<<gatesGrid, 256, 0, stream>>>(
        Xd + (size_t)t * BATCH * HID, h, W_ih_d, W_hh_d, G);
    lstm_cell_kernel<<<cellGrid, 256, 0, stream>>>(
        G, h, c, b_ih_d, b_hh_d, nullptr, h_bf);
    logits_wmma_kernel<<<logitGrid, 256, 0, stream>>>(h_bf, Wout_bf, logits);
    row_loss_kernel<<<BATCH, 256, 0, stream>>>(
        logits, b_out, target_lines + (t + 1) * BATCH, loss_part + t * BATCH);
  }

  final_loss_kernel<<<1, 256, 0, stream>>>(loss_part, (float*)d_out);
}
